// Model_14817637171453
// MI455X (gfx1250) — matrix operand order repaired and checked by reading, not compiled, not run on hardware
//
#include <hip/hip_runtime.h>
#include <cstdint>

typedef _Float16 v16h __attribute__((ext_vector_type(16)));
typedef _Float16 v8h  __attribute__((ext_vector_type(8)));
typedef _Float16 v4h  __attribute__((ext_vector_type(4)));
typedef float    v8f  __attribute__((ext_vector_type(8)));
typedef float    v4f  __attribute__((ext_vector_type(4)));

#define B_    2048
#define I_    128
#define H_    256
#define N_TOT 512          // H (m1 cols) + H (m2 cols)
#define K_TOT 131072       // 2H * 2I
#define K2    256          // K-chunk per j (= 2I)
#define MT_   128          // workgroup M tile
#define NT_   64           // workgroup N tile
#define KSPLIT 2
#define WROW  528          // padded W-tile LDS row stride (bytes): 256 halves + 8 pad
#define XROW  528          // padded x-tile LDS row stride (bytes)
#define XTILE_BYTES (MT_ * XROW)          // 67584
#define WTILE_BYTES (NT_ * WROW)          // 33792

static __device__ inline v16h cat16(v8h a, v8h b) {
  return __builtin_shufflevector(a, b, 0,1,2,3,4,5,6,7,8,9,10,11,12,13,14,15);
}

// ---------------- Kernel 1: W1||W2 f32 -> f16 into workspace -----------------
__global__ __launch_bounds__(256) void wconvert(const float* __restrict__ W1,
                                                const float* __restrict__ W2,
                                                _Float16* __restrict__ Wc) {
  const long half = (long)H_ * K_TOT;                 // 33,554,432
  long i = ((long)blockIdx.x * 256 + threadIdx.x) * 8;
  const float* src = (i < half) ? (W1 + i) : (W2 + (i - half));
  v4f a = *(const v4f*)src;
  v4f b = *(const v4f*)(src + 4);
  v8h o;
  o[0]=(_Float16)a[0]; o[1]=(_Float16)a[1]; o[2]=(_Float16)a[2]; o[3]=(_Float16)a[3];
  o[4]=(_Float16)b[0]; o[5]=(_Float16)b[1]; o[6]=(_Float16)b[2]; o[7]=(_Float16)b[3];
  *(v8h*)(Wc + i) = o;
}

// ---------------- Kernel 2: main WMMA GEMM with async-LDS W staging ----------
__global__ __launch_bounds__(256) void tgru_gemm(
    const float* __restrict__ x0, const float* __restrict__ x1,
    const float* __restrict__ h0, const float* __restrict__ h1,
    const _Float16* __restrict__ Wc, float* __restrict__ mpart)
{
  // [0, XTILE) x-tile | [XTILE, +WTILE) W buf0 | [XTILE+WTILE, +WTILE) W buf1
  __shared__ char smem[XTILE_BYTES + 2 * WTILE_BYTES];  // 132 KB -> 2 WG/WGP

  const int bid  = blockIdx.x;
  const int ks   = bid & 1;            // K split (j-half)
  const int nt   = (bid >> 1) & 7;     // N tile (64 cols)
  const int mt   = bid >> 4;           // M tile (128 rows)
  const int t    = threadIdx.x;
  const int lane = t & 31;
  const int wave = t >> 5;
  const int mw   = wave & 3;           // 4 waves along M (32 rows each)
  const int nw   = wave >> 2;          // 2 waves along N (32 cols each)
  const int hi   = lane >> 4;
  const int ln   = lane & 15;

  // ---- stage x tile f32->f16 into LDS: rows [mt*128, +128), 256 halves/row
  {
    int r  = t & 127;
    int ch = (t >> 7) * 128;           // 0 -> x0 half, 128 -> x1 half
    const float* src = (ch == 0 ? x0 : x1) + (mt * MT_ + r) * I_;
    #pragma unroll
    for (int c = 0; c < 128; c += 4) {
      v4f v = *(const v4f*)(src + c);
      v4h o;
      o[0]=(_Float16)v[0]; o[1]=(_Float16)v[1]; o[2]=(_Float16)v[2]; o[3]=(_Float16)v[3];
      *(v4h*)(smem + r * XROW + (ch + c) * 2) = o;
    }
  }

  const uint64_t wbase    = (uint64_t)(uintptr_t)Wc;
  const uint32_t smemBase = (uint32_t)(uintptr_t)&smem[0];   // LDS byte offset for asm
  const int row = t >> 2;              // 0..63  (W tile row = output col)
  const int ck  = t & 3;               // 4 lanes per row, 16B chunks
  // async issue of one W tile (64 rows x 512B) into LDS buffer at byte offset bufOff
  auto issue_w = [&](int bufOff, int j) {
    uint32_t lds0 = smemBase + (uint32_t)(bufOff + row * WROW + ck * 16);
    int g0 = (nt * NT_ + row) * (K_TOT * 2) + j * (K2 * 2) + ck * 16;
    #pragma unroll
    for (int p = 0; p < 8; ++p) {
      uint32_t la = lds0 + p * 64;
      int      ga = g0   + p * 64;
      asm volatile("global_load_async_to_lds_b128 %0, %1, %2"
                   :: "v"(la), "v"(ga), "s"(wbase) : "memory");
    }
  };

  v8f acc00 = {}, acc01 = {}, acc10 = {}, acc11 = {};
  const int jStart = ks * 256;
  const int jEnd   = jStart + 256;
  issue_w(XTILE_BYTES, jStart);
  int cur = 0;                          // 0 -> buf0, 1 -> buf1 (integer select only)

  const int rG0   = mt * MT_ + mw * 32 + ln;   // global row for mi=0 fragment
  const int xr0   = (mw * 32 + ln) * XROW;     // x-tile LDS row offsets (ints)
  const int xr1   = xr0 + 16 * XROW;
  const int wrRel = (nw * 32 + ln) * WROW;     // W-tile LDS row offset within buffer

  for (int jj = jStart; jj < jEnd; ++jj) {
    asm volatile("s_wait_asynccnt 0" ::: "memory");
    __syncthreads();
    if (jj + 1 < jEnd) issue_w(XTILE_BYTES + (cur ^ 1) * WTILE_BYTES, jj + 1);

    const float* sp = (jj < 256) ? h0 : h1;
    const int jc = jj & 255;
    _Float16 s0h = (_Float16)sp[(rG0)      * H_ + jc];
    _Float16 s1h = (_Float16)sp[(rG0 + 16) * H_ + jc];

    // integer offsets keep everything traceable to addrspace(3) -> ds_load_b128
    const int wb  = XTILE_BYTES + cur * WTILE_BYTES;
    const int wr0 = wb + wrRel;
    const int wr1 = wr0 + 16 * WROW;

    #pragma unroll
    for (int stp = 0; stp < 8; ++stp) {
      const int k2b  = stp * 32;
      const int aoff = (k2b + 8 * hi) * 2;     // A: K = {8hi..8hi+7, 16+8hi..}
      v8h a0lo = *(const v8h*)(smem + xr0 + aoff);
      v8h a0hi = *(const v8h*)(smem + xr0 + aoff + 32);
      v8h a1lo = *(const v8h*)(smem + xr1 + aoff);
      v8h a1hi = *(const v8h*)(smem + xr1 + aoff + 32);
      v16h a0 = cat16(a0lo, a0hi) * s0h;       // rank-1: x-frag scaled by s[b,j]
      v16h a1 = cat16(a1lo, a1hi) * s1h;

      const int boff = (k2b + 8 * hi) * 2;     // B: K = {8hi..8hi+7, 16+8hi..}
      v8h b0lo = *(const v8h*)(smem + wr0 + boff);
      v8h b0hi = *(const v8h*)(smem + wr0 + boff + 32);
      v8h b1lo = *(const v8h*)(smem + wr1 + boff);
      v8h b1hi = *(const v8h*)(smem + wr1 + boff + 32);
      v16h b0 = cat16(b0lo, b0hi);
      v16h b1 = cat16(b1lo, b1hi);

      acc00 = __builtin_amdgcn_wmma_f32_16x16x32_f16(false, a0, false, b0, (short)0, acc00, false, false);
      acc01 = __builtin_amdgcn_wmma_f32_16x16x32_f16(false, a0, false, b1, (short)0, acc01, false, false);
      acc10 = __builtin_amdgcn_wmma_f32_16x16x32_f16(false, a1, false, b0, (short)0, acc10, false, false);
      acc11 = __builtin_amdgcn_wmma_f32_16x16x32_f16(false, a1, false, b1, (short)0, acc11, false, false);
    }
    cur ^= 1;
  }

  // ---- write K-split partials (deterministic: separate buffer per split)
  float* mp = mpart + (long)ks * ((long)B_ * N_TOT);
  #pragma unroll
  for (int mi = 0; mi < 2; ++mi) {
    #pragma unroll
    for (int ni = 0; ni < 2; ++ni) {
      const v8f* a = (mi == 0) ? (ni == 0 ? &acc00 : &acc01)
                               : (ni == 0 ? &acc10 : &acc11);
      int nG = nt * NT_ + nw * 32 + ni * 16 + ln;
      #pragma unroll
      for (int vv = 0; vv < 8; ++vv) {
        int mG = mt * MT_ + mw * 32 + mi * 16 + vv + 8 * hi;  // C: M = vgpr + 8*hi
        mp[(long)mG * N_TOT + nG] = (*a)[vv];
      }
    }
  }
}

// ---------------- Kernel 3: epilogue (reduce splits, act, s@Wmerge) ----------
__global__ __launch_bounds__(256) void tgru_epi(
    const float* __restrict__ h0, const float* __restrict__ h1,
    const float* __restrict__ mpart, const float* __restrict__ b1,
    const float* __restrict__ b2, const float* __restrict__ Wm,
    float* __restrict__ out)
{
  __shared__ float sS[8][512];
  const int t  = threadIdx.x;
  const int rB = blockIdx.x * 8;
  #pragma unroll
  for (int i = 0; i < 16; ++i) {
    int idx = t + i * 256;
    int r = idx >> 9, j = idx & 511;
    sS[r][j] = (j < 256) ? h0[(rB + r) * H_ + j] : h1[(rB + r) * H_ + (j - 256)];
  }
  __syncthreads();
  const int l = t;
  float acc[8] = {};
  for (int j = 0; j < 512; ++j) {
    float w = Wm[j * H_ + l];
    #pragma unroll
    for (int r = 0; r < 8; ++r) acc[r] += sS[r][j] * w;
  }
  const float* p0 = mpart;
  const float* p1 = mpart + (long)B_ * N_TOT;
  float bb1 = b1[l], bb2 = b2[l];
  #pragma unroll
  for (int r = 0; r < 8; ++r) {
    int b = rB + r;
    long base = (long)b * N_TOT;
    float m1 = p0[base + l]       + p1[base + l];
    float m2 = p0[base + 256 + l] + p1[base + 256 + l];
    float st = tanhf(m1 + bb1);
    float u  = 1.0f / (1.0f + __expf(-(m2 + bb2)));
    float ns = u * st + (1.0f - u) * acc[r];
    out[(long)b * H_ + l] = ns;
    out[(long)B_ * H_ + (long)b * H_ + l] = ns;
  }
}

extern "C" void kernel_launch(void* const* d_in, const int* in_sizes, int n_in,
                              void* d_out, int out_size, void* d_ws, size_t ws_size,
                              hipStream_t stream) {
  (void)in_sizes; (void)n_in; (void)out_size; (void)ws_size;
  const float* x0 = (const float*)d_in[0];
  const float* x1 = (const float*)d_in[1];
  const float* h0 = (const float*)d_in[2];
  const float* h1 = (const float*)d_in[3];
  const float* W1 = (const float*)d_in[4];
  const float* b1 = (const float*)d_in[5];
  const float* W2 = (const float*)d_in[6];
  const float* b2 = (const float*)d_in[7];
  const float* Wm = (const float*)d_in[8];
  float* out = (float*)d_out;

  _Float16* Wc   = (_Float16*)d_ws;                                   // 134.2 MB
  float*    mprt = (float*)((char*)d_ws + (size_t)N_TOT * K_TOT * 2); // 2 x 4 MB

  wconvert<<<(N_TOT * (K_TOT / 8)) / 256, 256, 0, stream>>>(W1, W2, Wc); // 32768 blocks
  tgru_gemm<<<16 * 8 * KSPLIT, 256, 0, stream>>>(x0, x1, h0, h1, Wc, mprt);
  tgru_epi<<<B_ / 8, 256, 0, stream>>>(h0, h1, mprt, b1, b2, Wm, out);
}
